// SparseBasicBlock_69965017252009
// MI455X (gfx1250) — matrix operand order repaired and checked by reading, not compiled, not run on hardware
//
#include <hip/hip_runtime.h>

typedef __attribute__((ext_vector_type(16))) _Float16 v16h;
typedef __attribute__((ext_vector_type(8)))  _Float16 v8h;
typedef __attribute__((ext_vector_type(8)))  float    v8f;
typedef __attribute__((ext_vector_type(4)))  unsigned int u32x4;
typedef __attribute__((ext_vector_type(8)))  int i32x8;
typedef __attribute__((ext_vector_type(4)))  int i32x4;

// ---------------------------------------------------------------------------
// Pass 1: linear index per voxel (sorted ascending by construction).
// ---------------------------------------------------------------------------
__global__ void lin_kernel(const int* __restrict__ idx4, long long* __restrict__ lin,
                           const int* __restrict__ sT, const int* __restrict__ sH,
                           const int* __restrict__ sW, int n) {
  int i = blockIdx.x * blockDim.x + threadIdx.x;
  if (i >= n) return;
  long long T = sT[0], H = sH[0], W = sW[0];
  long long b = idx4[i * 4 + 0], t = idx4[i * 4 + 1];
  long long y = idx4[i * 4 + 2], x = idx4[i * 4 + 3];
  lin[i] = ((b * T + t) * H + y) * W + x;
}

// ---------------------------------------------------------------------------
// Pass 2: features fp32 -> f16, plus one trailing all-zero row (index n) that
// missing neighbors gather from. Also zero the mid buffer's zero row.
// ---------------------------------------------------------------------------
__global__ void cvt_kernel(const float* __restrict__ f, _Float16* __restrict__ fa,
                           _Float16* __restrict__ fm, int n) {
  int gid = blockIdx.x * blockDim.x + threadIdx.x;
  if (gid >= (n + 1) * 64) return;
  int row = gid >> 6;
  float v = (row < n) ? f[gid] : 0.f;
  fa[gid] = (_Float16)v;
  if (row >= n) fm[gid] = (_Float16)0.f;
}

// ---------------------------------------------------------------------------
// Pass 3: pre-shuffle both 27x64x64 weight tensors into the WMMA B-fragment
// lane layout: element ((off*2+kb)*4+nt)*512 + lane*16 + 2j holds
// w[K0][ncol], w[K0+1][ncol] with K0 = kb*32 + (lane>=16)*16 + 2j,
// ncol = nt*16 + (lane&15). Per offset this is one contiguous 8KB block that
// the conv kernel DMA-stages into LDS with the Tensor Data Mover.
// ---------------------------------------------------------------------------
__global__ void pack_w_kernel(const float* __restrict__ w1, const float* __restrict__ w2,
                              _Float16* __restrict__ wpk) {
  int gid = blockIdx.x * blockDim.x + threadIdx.x;
  const int per_conv = 27 * 2 * 4 * 32 * 8;
  if (gid >= 2 * per_conv) return;
  int conv = gid / per_conv;
  int r = gid % per_conv;
  int off = r / (2 * 4 * 32 * 8);  r %= 2 * 4 * 32 * 8;
  int kb  = r / (4 * 32 * 8);      r %= 4 * 32 * 8;
  int nt  = r / (32 * 8);          r %= 32 * 8;
  int lane = r / 8;
  int j    = r % 8;
  int laneL = lane & 15, laneH = lane >> 4;
  int K0   = kb * 32 + laneH * 16 + 2 * j;
  int ncol = nt * 16 + laneL;
  const float* wb = (conv ? w2 : w1) + off * 4096;
  size_t e = (size_t)conv * 27 * 4096 + (size_t)((off * 2 + kb) * 4 + nt) * 512
           + lane * 16 + 2 * j;
  wpk[e]     = (_Float16)wb[K0 * 64 + ncol];
  wpk[e + 1] = (_Float16)wb[(K0 + 1) * 64 + ncol];
}

// ---------------------------------------------------------------------------
// Pass 4: neighbor table nb[k][i] via binary search on sorted lin.
// Missing neighbor -> index n (the zero row). Shared by both convs.
// ---------------------------------------------------------------------------
__global__ void nb_kernel(const long long* __restrict__ lin, const int* __restrict__ idx4,
                          int* __restrict__ nb, const int* __restrict__ sT,
                          const int* __restrict__ sH, const int* __restrict__ sW, int n) {
  long long gid = (long long)blockIdx.x * blockDim.x + threadIdx.x;
  if (gid >= (long long)n * 27) return;
  int i = (int)(gid % n);
  int k = (int)(gid / n);
  int T = sT[0], H = sH[0], W = sW[0];
  int dt = k / 9 - 1, dy = (k / 3) % 3 - 1, dx = k % 3 - 1;
  int t = idx4[i * 4 + 1] + dt;
  int y = idx4[i * 4 + 2] + dy;
  int x = idx4[i * 4 + 3] + dx;
  int res = n;  // zero row
  if (t >= 0 && t < T && y >= 0 && y < H && x >= 0 && x < W) {
    long long target = lin[i] + (long long)dt * H * W + (long long)dy * W + dx;
    int lo = 0, hi = n;
    while (lo < hi) {
      int mid = (lo + hi) >> 1;
      if (lin[mid] < target) lo = mid + 1; else hi = mid;
    }
    if (lo < n && lin[lo] == target) res = lo;
  }
  nb[(size_t)k * n + i] = res;
}

// ---------------------------------------------------------------------------
// TDM: DMA one 8KB packed-weight block (2048 dwords, 1-D tile) into LDS.
// D# group0: count=1 | lds_addr | 57-bit global addr | type=2.
// D# group1: data_size=4B, tensor_dim0=tile_dim0=2048, dim1=1, stride=2048.
// ---------------------------------------------------------------------------
__device__ __forceinline__ void tdm_load_w8k(const _Float16* gsrc, unsigned lds_byte) {
  unsigned long long ga = (unsigned long long)(uintptr_t)gsrc;
  u32x4 g0;
  g0.x = 1u;                                            // count=1 (valid user D#)
  g0.y = lds_byte;                                      // lds_addr
  g0.z = (unsigned)(ga & 0xFFFFFFFFu);                  // global_addr[31:0]
  g0.w = (unsigned)((ga >> 32) & 0x01FFFFFFu) | (2u << 30);  // addr[56:32] | type=2
  i32x8 g1;
  g1[0] = 0x00020000;          // workgroup_mask=0, data_size=2 (4 bytes)
  g1[1] = (int)(2048u << 16);  // tensor_dim0[15:0] @ bits 63:48
  g1[2] = 0x00010000;          // tensor_dim1=1 @ bits 95:80
  g1[3] = (int)(2048u << 16);  // tile_dim0=2048 @ bits 127:112
  g1[4] = 1;                   // tile_dim1=1 @ bits 143:128
  g1[5] = 2048;                // tensor_dim0_stride[31:0]
  g1[6] = 0;
  g1[7] = 0;
  i32x4 z4 = {0, 0, 0, 0};
#if defined(__clang_major__) && (__clang_major__ >= 23)
  i32x8 z8 = {0, 0, 0, 0, 0, 0, 0, 0};
  __builtin_amdgcn_tensor_load_to_lds(g0, g1, z4, z4, z8, 0);
#else
  __builtin_amdgcn_tensor_load_to_lds(g0, g1, z4, z4, 0);
#endif
}

// ---------------------------------------------------------------------------
// Pass 5/6: gathered WMMA GEMM (16 voxels x 64 cout per wave, K = 27*64).
// Weights are TDM-staged into a double-buffered LDS slab (8KB per offset);
// B fragments come from LDS (ds_load_b128), A rows are gathered from global
// (L2-resident). Fused LayerNorm epilogue through a padded LDS tile.
//   mode 0: out = f16( relu(LN(conv)) )       -> mid buffer for conv2
//   mode 1: out = f32( relu(LN(conv)+resid) ) -> d_out
// EXEC stays all-ones through the WMMA loop (wave-uniform branches only).
// ---------------------------------------------------------------------------
__global__ __launch_bounds__(128)
void conv_kernel(const _Float16* __restrict__ feat,   // (n+1) x 64 f16
                 const int* __restrict__ nb,          // [27][n]
                 const _Float16* __restrict__ wpk,    // packed B fragments, 27*4096 f16
                 const float* __restrict__ gamma, const float* __restrict__ beta,
                 const float* __restrict__ resid,
                 _Float16* __restrict__ outh, float* __restrict__ outf,
                 int n, int mode) {
  __shared__ __align__(16) _Float16 smw[2][4096];  // double-buffered weight slab
  __shared__ float tile[64][65];                   // LN staging, bank-safe rows

  const int tid   = threadIdx.x;
  const int wave  = tid >> 5;
  const int lane  = tid & 31;
  const int laneL = lane & 15;
  const int laneH = lane >> 4;
  const int m0 = (blockIdx.x * 4 + wave) * 16;

  const v8f z = {0.f, 0.f, 0.f, 0.f, 0.f, 0.f, 0.f, 0.f};
  v8f acc[4] = {z, z, z, z};
  union AB { v16h v; v8h h[2]; };
  AB A, B0, B1, B2, B3;

  const unsigned lds_w0 = (unsigned)(uintptr_t)&smw[0][0];
  const unsigned lds_w1 = (unsigned)(uintptr_t)&smw[1][0];

  if (wave == 0) tdm_load_w8k(wpk, lds_w0);  // prime offset 0

  for (int off = 0; off < 27; ++off) {
    __builtin_amdgcn_s_wait_tensorcnt((short)0);  // issuing wave: data landed
    __syncthreads();                              // publish LDS slab; retire prev reads
    if (wave == 0 && off + 1 < 27)                // stream next offset's 8KB via TDM
      tdm_load_w8k(wpk + (size_t)(off + 1) * 4096, ((off + 1) & 1) ? lds_w1 : lds_w0);

    const int row = nb[(size_t)off * n + m0 + laneL];
    if (off + 1 < 27)  // prefetch next offset's neighbor indices (global_prefetch_b8)
      __builtin_prefetch(&nb[(size_t)(off + 1) * n + m0], 0, 3);
    const _Float16* arow = feat + (size_t)row * 64 + laneH * 16;
    const _Float16* lw   = &smw[off & 1][0] + lane * 16;

#pragma unroll
    for (int kb = 0; kb < 2; ++kb) {
      // A fragment in this code's k = 16h + s order: one contiguous 32B run of 16 k per lane (global gather)
      A.h[0] = *(const v8h*)(arow + kb * 32);
      A.h[1] = *(const v8h*)(arow + kb * 32 + 8);
      const _Float16* wk = lw + kb * 2048;
      // preload all four B fragments from LDS, then burst 4 WMMAs
      B0.h[0] = *(const v8h*)(wk);           B0.h[1] = *(const v8h*)(wk + 8);
      B1.h[0] = *(const v8h*)(wk + 512);     B1.h[1] = *(const v8h*)(wk + 520);
      B2.h[0] = *(const v8h*)(wk + 1024);    B2.h[1] = *(const v8h*)(wk + 1032);
      B3.h[0] = *(const v8h*)(wk + 1536);    B3.h[1] = *(const v8h*)(wk + 1544);
      acc[0] = __builtin_amdgcn_wmma_f32_16x16x32_f16(false, A.v, false, B0.v,
                                                      (short)0, acc[0], false, false);
      acc[1] = __builtin_amdgcn_wmma_f32_16x16x32_f16(false, A.v, false, B1.v,
                                                      (short)0, acc[1], false, false);
      acc[2] = __builtin_amdgcn_wmma_f32_16x16x32_f16(false, A.v, false, B2.v,
                                                      (short)0, acc[2], false, false);
      acc[3] = __builtin_amdgcn_wmma_f32_16x16x32_f16(false, A.v, false, B3.v,
                                                      (short)0, acc[3], false, false);
    }
  }

  // C/D layout: lane L holds (M = j + 8*laneH, N = nt*16 + laneL) in acc[nt][j]
#pragma unroll
  for (int nt = 0; nt < 4; ++nt)
#pragma unroll
    for (int j = 0; j < 8; ++j)
      tile[wave * 16 + 8 * laneH + j][nt * 16 + laneL] = acc[nt][j];
  __syncthreads();

  if (tid < 64) {
    const int r = tid;
    float s = 0.f;
#pragma unroll 8
    for (int c = 0; c < 64; ++c) s += tile[r][c];
    const float mean = s * 0.015625f;
    float var = 0.f;
#pragma unroll 8
    for (int c = 0; c < 64; ++c) { float d = tile[r][c] - mean; var += d * d; }
    const float rstd = __frsqrt_rn(var * 0.015625f + 1e-5f);
    const size_t gr = ((size_t)blockIdx.x * 64 + r) * 64;
    if (mode == 0) {
      for (int c = 0; c < 64; ++c) {
        float y = (tile[r][c] - mean) * rstd * gamma[c] + beta[c];
        outh[gr + c] = (_Float16)(y > 0.f ? y : 0.f);
      }
    } else {
      for (int c = 0; c < 64; ++c) {
        float y = (tile[r][c] - mean) * rstd * gamma[c] + beta[c] + resid[gr + c];
        outf[gr + c] = y > 0.f ? y : 0.f;
      }
    }
  }
}

// ---------------------------------------------------------------------------
extern "C" void kernel_launch(void* const* d_in, const int* in_sizes, int n_in,
                              void* d_out, int out_size, void* d_ws, size_t ws_size,
                              hipStream_t stream) {
  const float* features = (const float*)d_in[0];
  const int*   indices  = (const int*)d_in[1];
  const float* w1 = (const float*)d_in[2];
  const float* g1 = (const float*)d_in[3];
  const float* b1 = (const float*)d_in[4];
  const float* w2 = (const float*)d_in[5];
  const float* g2 = (const float*)d_in[6];
  const float* b2 = (const float*)d_in[7];
  const int* sT = (const int*)d_in[8];
  const int* sH = (const int*)d_in[9];
  const int* sW = (const int*)d_in[10];

  const int n = in_sizes[0] / 64;
  if (n <= 0 || (n & 63)) return;  // layout assumes 64-voxel blocks (400000 ok)

  // Workspace carve-out (256B aligned slabs)
  size_t cur = 0;
  auto take = [&](size_t bytes) -> void* {
    void* p = (char*)d_ws + cur;
    cur += (bytes + 255) & ~(size_t)255;
    return p;
  };
  long long* lin = (long long*)take((size_t)n * 8);
  int*       nb  = (int*)take((size_t)27 * n * 4);
  _Float16*  fa  = (_Float16*)take((size_t)(n + 1) * 64 * 2);
  _Float16*  fm  = (_Float16*)take((size_t)(n + 1) * 64 * 2);
  _Float16*  wpk = (_Float16*)take((size_t)2 * 27 * 4096 * 2);
  if (cur > ws_size) return;

  lin_kernel<<<(n + 255) / 256, 256, 0, stream>>>(indices, lin, sT, sH, sW, n);
  cvt_kernel<<<((n + 1) * 64 + 255) / 256, 256, 0, stream>>>(features, fa, fm, n);
  pack_w_kernel<<<(2 * 27 * 2 * 4 * 32 * 8 + 255) / 256, 256, 0, stream>>>(w1, w2, wpk);
  {
    long long tot = (long long)n * 27;
    nb_kernel<<<(unsigned)((tot + 255) / 256), 256, 0, stream>>>(lin, indices, nb,
                                                                 sT, sH, sW, n);
  }
  // conv1 -> LN -> ReLU -> f16 mid buffer
  conv_kernel<<<n / 64, 128, 0, stream>>>(fa, nb, wpk, g1, b1, features,
                                          fm, (float*)d_out, n, 0);
  // conv2 -> LN -> +residual -> ReLU -> fp32 output
  conv_kernel<<<n / 64, 128, 0, stream>>>(fm, nb, wpk + (size_t)27 * 4096, g2, b2,
                                          features, fm, (float*)d_out, n, 1);
}
